// MultiHeadAttentionLayer_55121610277464
// MI455X (gfx1250) — matrix operand order repaired and checked by reading, not compiled, not run on hardware
//
#include <hip/hip_runtime.h>
#include <hip/hip_bf16.h>
#include <stdint.h>

typedef __attribute__((ext_vector_type(16))) _Float16 v16h;
typedef __attribute__((ext_vector_type(8)))  float    v8f;

#define BB 2
#define SS 2048
#define HH 1024
#define NH 16
#define HD 64

union Frag16 { v16h v; _Float16 h[16]; uint32_t u[8]; };
union FragC  { v8f  v; float f[8]; };

// wave32 xor-shuffle via ds_bpermute (index bits [6:2] used -> wraps in wave)
__device__ __forceinline__ float bperm_xor(float x, int lane, int mask) {
    int addr = (lane ^ mask) << 2;
    return __int_as_float(__builtin_amdgcn_ds_bpermute(addr, __float_as_int(x)));
}

// CDNA5 async global->LDS copy (16B per lane), tracked by ASYNCcnt.
// LDS offset = low 32 bits of the flat shared-memory address (LDS aperture
// keeps the byte offset in addr[31:0]).
__device__ __forceinline__ void async_ld_b128(_Float16* lds_dst, const _Float16* src) {
    uint32_t ldsoff = (uint32_t)(uintptr_t)lds_dst;
    uint64_t gaddr  = (uint64_t)(uintptr_t)src;
    asm volatile("global_load_async_to_lds_b128 %0, %1, off"
                 :: "v"(ldsoff), "v"(gaddr) : "memory");
}
__device__ __forceinline__ void async_wait0() {
    asm volatile("s_wait_asynccnt 0x0" ::: "memory");
}

__global__ void f32_to_f16(const float* __restrict__ in, _Float16* __restrict__ out, int n) {
    int i = blockIdx.x * blockDim.x + threadIdx.x;
    if (i < n) out[i] = (_Float16)in[i];
}

// convert + transpose: in [R][C] f32 -> out [C][R] f16 (tiled through LDS)
__global__ __launch_bounds__(256)
void f32_to_f16_T(const float* __restrict__ in, _Float16* __restrict__ out, int R, int C) {
    __shared__ _Float16 tile[32][33];
    const int c0 = blockIdx.x * 32, r0 = blockIdx.y * 32;
    const int tx = threadIdx.x & 31, ty = threadIdx.x >> 5;   // 32 x 8
#pragma unroll
    for (int i = 0; i < 32; i += 8)
        tile[ty + i][tx] = (_Float16)in[(size_t)(r0 + ty + i) * C + c0 + tx];
    __syncthreads();
#pragma unroll
    for (int i = 0; i < 32; i += 8)
        out[(size_t)(c0 + ty + i) * R + r0 + tx] = tile[tx][ty + i];
}

// C = A(f16,[M,K]) * Bt(f16,[N,K] row-major, i.e. pre-transposed B) + bias.
// MODE 0: scatter f16 to [B,NH,S,HD]   (Q/K heads layout)
// MODE 2: scatter f16 to [B,NH,HD,S]   (V transposed for PV WMMA)
// MODE 1: f32 row-major [M,N]          (final output)
// Tile 128x64, 256 threads (8 waves, 32x32 each).  Tiles staged with
// async global->LDS (ASYNCcnt), fragments read as b128 from LDS.
template<int MODE>
__global__ __launch_bounds__(256)
void gemm_wmma(const _Float16* __restrict__ A, const _Float16* __restrict__ Bt,
               const float* __restrict__ bias, void* __restrict__ outp,
               int M, int N, int Kd)
{
    constexpr int TM = 128, TN = 64, TK = 32;
    constexpr int LDA = TK + 8;   // 40 halves
    __shared__ _Float16 sA[TM * LDA];
    __shared__ _Float16 sB[TN * LDA];   // Bt tile: [n][k]

    const int tid  = threadIdx.x;
    const int lane = tid & 31;
    const int w    = tid >> 5;           // 0..7
    const int wr   = w & 3, wc = w >> 2; // 4x2 wave grid
    const int g    = lane >> 4;          // half-wave
    const int nn   = lane & 15;
    const int m0   = blockIdx.y * TM;
    const int n0   = blockIdx.x * TN;

    // per-thread staging addresses (invariant across k-steps except k offset)
    const int arow = tid >> 1, aseg = tid & 1;
    const int brow = tid >> 2, bseg = tid & 3;
    const _Float16* aSrc = A  + (size_t)(m0 + arow) * Kd + aseg * 16;
    const _Float16* bSrc = Bt + (size_t)(n0 + brow) * Kd + bseg * 8;
    _Float16* aDst = sA + arow * LDA + aseg * 16;
    _Float16* bDst = sB + brow * LDA + bseg * 8;

    FragC acc[2][2];
#pragma unroll
    for (int i = 0; i < 2; i++)
#pragma unroll
        for (int j = 0; j < 2; j++)
#pragma unroll
            for (int r = 0; r < 8; r++) acc[i][j].f[r] = 0.f;

    for (int k0 = 0; k0 < Kd; k0 += TK) {
        // async-stage A (128x32) and B (64x32) tiles directly into LDS
        async_ld_b128(aDst,     aSrc + k0);
        async_ld_b128(aDst + 8, aSrc + k0 + 8);
        async_ld_b128(bDst,     bSrc + k0);
        async_wait0();
        __syncthreads();

        Frag16 af[2], bf[2];
#pragma unroll
        for (int mi = 0; mi < 2; mi++) {
            const _Float16* ap = sA + (wr * 32 + mi * 16 + nn) * LDA;
            *(float4*)&af[mi].h[0] = *(const float4*)(ap + g * 8);
            *(float4*)&af[mi].h[8] = *(const float4*)(ap + 16 + g * 8);
        }
#pragma unroll
        for (int ni = 0; ni < 2; ni++) {
            const _Float16* bp = sB + (wc * 32 + ni * 16 + nn) * LDA + g * 8;
            *(float4*)&bf[ni].h[0] = *(const float4*)(bp);
            *(float4*)&bf[ni].h[8] = *(const float4*)(bp + 16);
        }
#pragma unroll
        for (int mi = 0; mi < 2; mi++)
#pragma unroll
            for (int ni = 0; ni < 2; ni++)
                acc[mi][ni].v = __builtin_amdgcn_wmma_f32_16x16x32_f16(
                    false, af[mi].v, false, bf[ni].v, (short)0, acc[mi][ni].v, false, false);
        __syncthreads();
    }

#pragma unroll
    for (int mi = 0; mi < 2; mi++)
#pragma unroll
        for (int ni = 0; ni < 2; ni++)
#pragma unroll
            for (int r = 0; r < 8; r++) {
                int m = m0 + wr * 32 + mi * 16 + 8 * g + r;
                int n = n0 + wc * 32 + ni * 16 + nn;
                float val = acc[mi][ni].f[r] + bias[n];
                if (MODE == 0) {
                    _Float16* out = (_Float16*)outp;
                    int b = m >> 11, s = m & (SS - 1);
                    int head = n >> 6, d = n & (HD - 1);
                    out[((size_t)(b * NH + head) * SS + s) * HD + d] = (_Float16)val;
                } else if (MODE == 2) {
                    _Float16* out = (_Float16*)outp;
                    int b = m >> 11, s = m & (SS - 1);
                    int head = n >> 6, d = n & (HD - 1);
                    out[((size_t)(b * NH + head) * HD + d) * SS + s] = (_Float16)val;
                } else {
                    ((float*)outp)[(size_t)m * N + n] = val;
                }
            }
}

// Flash attention: Q/K in [B*NH, S, HD] f16, V in [B*NH, HD, S] f16 (transposed).
// f32 online softmax.  Block: 128 threads (4 waves), 64 query rows; 64-key chunks.
// K/V chunks staged with async global->LDS.
__global__ __launch_bounds__(128)
void attn_wmma(const _Float16* __restrict__ Qb, const _Float16* __restrict__ Kb,
               const _Float16* __restrict__ Vtb, const int* __restrict__ amask,
               _Float16* __restrict__ ctx)
{
    constexpr int LD = HD + 8;   // 72 halves
    __shared__ _Float16 sQ[64 * LD];
    __shared__ _Float16 sK[64 * LD];       // [key][d]
    __shared__ _Float16 sVt[64 * LD];      // [d][key]
    __shared__ _Float16 sP[4 * 16 * LD];   // per-wave P staging [row][key]

    const int tid  = threadIdx.x;
    const int lane = tid & 31;
    const int w    = tid >> 5;     // 0..3
    const int g    = lane >> 4;
    const int nn   = lane & 15;

    const int bh = blockIdx.y;     // b*NH + h
    const int b  = bh >> 4;
    const int h  = bh & 15;
    const int q0 = blockIdx.x * 64;

    const _Float16* Qp  = Qb  + (size_t)bh * SS * HD;
    const _Float16* Kp  = Kb  + (size_t)bh * SS * HD;
    const _Float16* Vtp = Vtb + (size_t)bh * HD * SS;
    const int* mrow = amask + b * SS;

    const int row = tid >> 1, seg = tid & 1;   // staging coords (64 rows x 2 segs)

    // stage Q tile 64x64 (async)
    {
        const _Float16* src = Qp + (size_t)(q0 + row) * HD + seg * 32;
        _Float16* dst = sQ + row * LD + seg * 32;
#pragma unroll
        for (int j = 0; j < 32; j += 8)
            async_ld_b128(dst + j, src + j);
    }
    async_wait0();
    __syncthreads();

    Frag16 qf[2];
#pragma unroll
    for (int ks = 0; ks < 2; ks++) {
        const _Float16* qp = sQ + (w * 16 + nn) * LD + ks * 32;
        *(float4*)&qf[ks].h[0] = *(const float4*)(qp + g * 8);
        *(float4*)&qf[ks].h[8] = *(const float4*)(qp + 16 + g * 8);
    }

    FragC oacc[4];
#pragma unroll
    for (int t = 0; t < 4; t++)
#pragma unroll
        for (int r = 0; r < 8; r++) oacc[t].f[r] = 0.f;
    float mi[8], li[8];
#pragma unroll
    for (int r = 0; r < 8; r++) { mi[r] = -1e30f; li[r] = 0.f; }

    for (int c = 0; c < SS / 64; c++) {
        const int key0 = c * 64;
        __syncthreads();   // all waves done reading previous sK/sVt
        {
            const _Float16* srcK = Kp  + (size_t)(key0 + row) * HD + seg * 32;
            const _Float16* srcV = Vtp + (size_t)row * SS + key0 + seg * 32;  // row = d
            _Float16* dK = sK  + row * LD + seg * 32;
            _Float16* dV = sVt + row * LD + seg * 32;
#pragma unroll
            for (int j = 0; j < 32; j += 8) {
                async_ld_b128(dK + j, srcK + j);
                async_ld_b128(dV + j, srcV + j);
            }
        }
        async_wait0();
        __syncthreads();

        // S = Q K^T : WMMA-K = head dim; B[d][key] = sK[key][d], pairs contiguous
        FragC sacc[4];
#pragma unroll
        for (int t = 0; t < 4; t++)
#pragma unroll
            for (int r = 0; r < 8; r++) sacc[t].f[r] = 0.f;
#pragma unroll
        for (int ks = 0; ks < 2; ks++)
#pragma unroll
            for (int t = 0; t < 4; t++) {
                Frag16 bk;
                const _Float16* kp = sK + (t * 16 + nn) * LD + ks * 32 + g * 8;
                *(float4*)&bk.h[0] = *(const float4*)(kp);
                *(float4*)&bk.h[8] = *(const float4*)(kp + 16);
                sacc[t].v = __builtin_amdgcn_wmma_f32_16x16x32_f16(
                    false, qf[ks].v, false, bk.v, (short)0, sacc[t].v, false, false);
            }

        float madd[4];
#pragma unroll
        for (int t = 0; t < 4; t++) {
            int key = key0 + t * 16 + nn;
            madd[t] = (1.0f - (float)mrow[key]) * -10000.0f;
        }
        float sreg[4][8];
#pragma unroll
        for (int t = 0; t < 4; t++)
#pragma unroll
            for (int r = 0; r < 8; r++)
                sreg[t][r] = sacc[t].f[r] * 0.125f + madd[t];

        // online softmax; rows m = 8g + r spread over a 16-lane group
#pragma unroll
        for (int r = 0; r < 8; r++) {
            float cm = fmaxf(fmaxf(sreg[0][r], sreg[1][r]), fmaxf(sreg[2][r], sreg[3][r]));
            cm = fmaxf(cm, bperm_xor(cm, lane, 1));
            cm = fmaxf(cm, bperm_xor(cm, lane, 2));
            cm = fmaxf(cm, bperm_xor(cm, lane, 4));
            cm = fmaxf(cm, bperm_xor(cm, lane, 8));
            float mnew = fmaxf(mi[r], cm);
            float corr = __expf(mi[r] - mnew);
            float ps = 0.f;
#pragma unroll
            for (int t = 0; t < 4; t++) {
                float p = __expf(sreg[t][r] - mnew);
                sreg[t][r] = p;
                ps += p;
            }
            ps += bperm_xor(ps, lane, 1);
            ps += bperm_xor(ps, lane, 2);
            ps += bperm_xor(ps, lane, 4);
            ps += bperm_xor(ps, lane, 8);
            li[r] = li[r] * corr + ps;
            mi[r] = mnew;
#pragma unroll
            for (int t = 0; t < 4; t++) oacc[t].f[r] *= corr;
        }

        // C-layout -> A-layout via per-wave LDS (same-wave DS ops are in-order)
        _Float16* pw = sP + w * 16 * LD;
#pragma unroll
        for (int t = 0; t < 4; t++)
#pragma unroll
            for (int r = 0; r < 8; r++)
                pw[(8 * g + r) * LD + t * 16 + nn] = (_Float16)sreg[t][r];

        // O += P V : WMMA-K = key; B[key][d] = sVt[d][key], pairs contiguous
#pragma unroll
        for (int ks = 0; ks < 2; ks++) {
            Frag16 ap;
            const _Float16* pp = pw + nn * LD + ks * 32;
            *(float4*)&ap.h[0] = *(const float4*)(pp + g * 8);
            *(float4*)&ap.h[8] = *(const float4*)(pp + 16 + g * 8);
#pragma unroll
            for (int dt = 0; dt < 4; dt++) {
                Frag16 bv;
                const _Float16* vp = sVt + (dt * 16 + nn) * LD + ks * 32 + g * 8;
                *(float4*)&bv.h[0] = *(const float4*)(vp);
                *(float4*)&bv.h[8] = *(const float4*)(vp + 16);
                oacc[dt].v = __builtin_amdgcn_wmma_f32_16x16x32_f16(
                    false, ap.v, false, bv.v, (short)0, oacc[dt].v, false, false);
            }
        }
    }

    // write ctx (f16) back in [B,S,H] with head offset
#pragma unroll
    for (int dt = 0; dt < 4; dt++)
#pragma unroll
        for (int r = 0; r < 8; r++) {
            int q   = q0 + w * 16 + 8 * g + r;
            int col = h * HD + dt * 16 + nn;
            float o = oacc[dt].f[r] / li[r];
            ctx[(size_t)(b * SS + q) * HH + col] = (_Float16)o;
        }
}

extern "C" void kernel_launch(void* const* d_in, const int* in_sizes, int n_in,
                              void* d_out, int out_size, void* d_ws, size_t ws_size,
                              hipStream_t stream)
{
    const float* X  = (const float*)d_in[0];
    const int*   AM = (const int*)  d_in[1];
    const float* Wq = (const float*)d_in[2];
    const float* bq = (const float*)d_in[3];
    const float* Wk = (const float*)d_in[4];
    const float* bk = (const float*)d_in[5];
    const float* Wv = (const float*)d_in[6];
    const float* bv = (const float*)d_in[7];
    const float* Wo = (const float*)d_in[8];
    const float* bo = (const float*)d_in[9];
    float* out = (float*)d_out;

    const size_t NX = (size_t)BB * SS * HH;   // 4M elems
    const size_t NW = (size_t)HH * HH;        // 1M elems
    _Float16* Xh  = (_Float16*)d_ws;
    _Float16* Wqt = Xh  + NX;     // [N][K] transposed f16 weights
    _Float16* Wkt = Wqt + NW;
    _Float16* Wvt = Wkt + NW;
    _Float16* Wot = Wvt + NW;
    _Float16* Qh  = Wot + NW;     // [B,NH,S,HD]
    _Float16* Kh  = Qh  + NX;     // [B,NH,S,HD]
    _Float16* Vt  = Kh  + NX;     // [B,NH,HD,S]
    _Float16* Ch  = Vt  + NX;     // ctx f16 [B,S,H]   (total 48 MiB)

    f32_to_f16<<<dim3((unsigned)((NX + 255) / 256)), 256, 0, stream>>>(X, Xh, (int)NX);
    dim3 gt(HH / 32, HH / 32);
    f32_to_f16_T<<<gt, 256, 0, stream>>>(Wq, Wqt, HH, HH);
    f32_to_f16_T<<<gt, 256, 0, stream>>>(Wk, Wkt, HH, HH);
    f32_to_f16_T<<<gt, 256, 0, stream>>>(Wv, Wvt, HH, HH);
    f32_to_f16_T<<<gt, 256, 0, stream>>>(Wo, Wot, HH, HH);

    dim3 gg(HH / 64, (BB * SS) / 128);   // (16, 32)
    gemm_wmma<0><<<gg, 256, 0, stream>>>(Xh, Wqt, bq, Qh, BB * SS, HH, HH);
    gemm_wmma<0><<<gg, 256, 0, stream>>>(Xh, Wkt, bk, Kh, BB * SS, HH, HH);
    gemm_wmma<2><<<gg, 256, 0, stream>>>(Xh, Wvt, bv, Vt, BB * SS, HH, HH);

    dim3 ga(SS / 64, BB * NH);           // (32, 32)
    attn_wmma<<<ga, 128, 0, stream>>>(Qh, Kh, Vt, AM, Ch);

    gemm_wmma<1><<<gg, 256, 0, stream>>>(Ch, Wot, bo, out, BB * SS, HH, HH);
}
